// SRGC_13975823582059
// MI455X (gfx1250) — matrix operand order repaired and checked by reading, not compiled, not run on hardware
//
#include <hip/hip_runtime.h>
#include <hip/hip_bf16.h>

typedef __attribute__((ext_vector_type(16))) _Float16 v16h;
typedef __attribute__((ext_vector_type(8)))  float    v8f;

#define IN_F  128
#define OUT_F 128   // NUM_HEADS * OUT_FEATS
#define NHEAD 4
#define NETYPE 8

// ---------------------------------------------------------------------------
// 1) ft = feat @ W^T via v_wmma_f32_16x16x32_f16.
//    5 waves / block, each wave owns a 16-row tile and all 128 output cols.
// ---------------------------------------------------------------------------
__global__ __launch_bounds__(160) void srgc_fc_wmma(
    const float* __restrict__ feat, const float* __restrict__ W,
    float* __restrict__ ft, int n_nodes) {
  __shared__ __align__(32) _Float16 wlds[OUT_F * IN_F];  // 32KB, W[n][k] as f16
  for (int i = threadIdx.x; i < OUT_F * IN_F; i += blockDim.x)
    wlds[i] = (_Float16)W[i];
  __syncthreads();

  const int wave = threadIdx.x >> 5;
  const int lane = threadIdx.x & 31;
  const int tiles = n_nodes >> 4;                 // n_nodes % 16 == 0 (100000)
  const int tile = blockIdx.x * 5 + wave;
  if (tile >= tiles) return;                      // wave-uniform; after barrier

  const int row0 = tile * 16;
  const int arow = row0 + (lane & 15);
  const int koff = (lane < 16) ? 0 : 8;           // 16-bit A 16x32 layout
  const int col  = lane & 15;
  const int kb   = (lane < 16) ? 0 : 8;           // B 32x16 layout
  const float* __restrict__ aptr = feat + (size_t)arow * IN_F;

  v8f acc[8] = {};
  for (int ks = 0; ks < IN_F; ks += 32) {
    // A fragment: K = {koff..koff+7, koff+16..koff+23} of this K-chunk
    const float* p = aptr + ks + koff;
    float4 f0 = *(const float4*)(p);
    float4 f1 = *(const float4*)(p + 4);
    float4 f2 = *(const float4*)(p + 16);
    float4 f3 = *(const float4*)(p + 20);
    v16h a;
    a[0]=(_Float16)f0.x; a[1]=(_Float16)f0.y; a[2]=(_Float16)f0.z; a[3]=(_Float16)f0.w;
    a[4]=(_Float16)f1.x; a[5]=(_Float16)f1.y; a[6]=(_Float16)f1.z; a[7]=(_Float16)f1.w;
    a[8]=(_Float16)f2.x; a[9]=(_Float16)f2.y; a[10]=(_Float16)f2.z; a[11]=(_Float16)f2.w;
    a[12]=(_Float16)f3.x; a[13]=(_Float16)f3.y; a[14]=(_Float16)f3.z; a[15]=(_Float16)f3.w;
#pragma unroll
    for (int nt = 0; nt < 8; ++nt) {
      const int n = nt * 16 + col;
      typedef __attribute__((ext_vector_type(8))) _Float16 v8h_;
      v8h_ b0 = *(const v8h_*)(&wlds[n * IN_F + ks + kb]);
      v8h_ b1 = *(const v8h_*)(&wlds[n * IN_F + ks + kb + 16]);
      v16h b = __builtin_shufflevector(b0, b1, 0,1,2,3,4,5,6,7,8,9,10,11,12,13,14,15);
      acc[nt] = __builtin_amdgcn_wmma_f32_16x16x32_f16(
          false, a, false, b, (short)0, acc[nt], false, false);
    }
  }
  // D layout: VGPR r -> M = r (lanes 0-15) or 8+r (lanes 16-31), N = nt*16+col
  const int mrow = row0 + ((lane >= 16) ? 8 : 0);
#pragma unroll
  for (int nt = 0; nt < 8; ++nt)
#pragma unroll
    for (int r = 0; r < 8; ++r)
      ft[(size_t)(mrow + r) * OUT_F + nt * 16 + col] = acc[nt][r];
}

// ---------------------------------------------------------------------------
// 2) per-destination etype histogram (1 int atomic per edge)
// ---------------------------------------------------------------------------
__global__ void srgc_count(const int* __restrict__ e_feat,
                           const int* __restrict__ dst,
                           int* __restrict__ cnt, int E) {
  int e = blockIdx.x * blockDim.x + threadIdx.x;
  if (e < E) atomicAdd(&cnt[(size_t)dst[e] * NETYPE + e_feat[e]], 1);
}

// ---------------------------------------------------------------------------
// 3) per-node softmax stats: m = max over present types, inv = 1/sum
// ---------------------------------------------------------------------------
__global__ void srgc_node_stats(const int* __restrict__ cnt,
                                const float* __restrict__ emb, /* (8,4) */
                                float* __restrict__ mbuf, float* __restrict__ invbuf,
                                int N) {
  int n = blockIdx.x * blockDim.x + threadIdx.x;
  if (n >= N) return;
  int c[NETYPE];
#pragma unroll
  for (int t = 0; t < NETYPE; ++t) c[t] = cnt[(size_t)n * NETYPE + t];
#pragma unroll
  for (int h = 0; h < NHEAD; ++h) {
    float m = -1e30f;
#pragma unroll
    for (int t = 0; t < NETYPE; ++t)
      if (c[t] > 0) m = fmaxf(m, emb[t * NHEAD + h]);
    float s = 0.f;
#pragma unroll
    for (int t = 0; t < NETYPE; ++t)
      if (c[t] > 0) s += (float)c[t] * __expf(emb[t * NHEAD + h] - m);
    mbuf[(size_t)n * NHEAD + h]   = (m == -1e30f) ? 0.f : m;
    invbuf[(size_t)n * NHEAD + h] = (s > 0.f) ? (1.f / s) : 0.f;
  }
}

// ---------------------------------------------------------------------------
// 4) per-edge attention, broadcast over drop_blocks into d_out attn slab
// ---------------------------------------------------------------------------
__global__ void srgc_edge_attn(const int* __restrict__ e_feat,
                               const int* __restrict__ dst,
                               const float* __restrict__ emb,
                               const float* __restrict__ mbuf,
                               const float* __restrict__ invbuf,
                               float* __restrict__ attn_out, int E, int DB) {
  int e = blockIdx.x * blockDim.x + threadIdx.x;
  if (e >= E) return;
  const int et = e_feat[e], d = dst[e];
  float* o = attn_out + (size_t)e * NHEAD * DB;
#pragma unroll
  for (int h = 0; h < NHEAD; ++h) {
    float a = __expf(emb[et * NHEAD + h] - mbuf[(size_t)d * NHEAD + h]) *
              invbuf[(size_t)d * NHEAD + h];
    for (int b = 0; b < DB; ++b) o[h * DB + b] = a;
  }
}

// ---------------------------------------------------------------------------
// 5) aggregation: one wave per edge; lane owns 4 consecutive of the 128
//    components (same head). ft gather is L2-resident; f32 atomics into rst.
// ---------------------------------------------------------------------------
__global__ __launch_bounds__(256) void srgc_aggregate(
    const float* __restrict__ ft, const int* __restrict__ src,
    const int* __restrict__ dst, const float* __restrict__ attn_out,
    float* __restrict__ rst, int E, int DB) {
  const int gwave = (blockIdx.x * blockDim.x + threadIdx.x) >> 5;
  const int lane  = threadIdx.x & 31;
  if (gwave >= E) return;
  const int s = src[gwave], d = dst[gwave];
  const int c0 = lane * 4;            // component 0..127, all 4 in same head
  const int h  = c0 >> 5;
  const float a = attn_out[(size_t)gwave * NHEAD * DB + h * DB];
  const float4 f = *(const float4*)(ft + (size_t)s * OUT_F + c0);
  float* o = rst + (size_t)d * OUT_F + c0;
  atomicAdd(o + 0, f.x * a);
  atomicAdd(o + 1, f.y * a);
  atomicAdd(o + 2, f.z * a);
  atomicAdd(o + 3, f.w * a);
}

extern "C" void kernel_launch(void* const* d_in, const int* in_sizes, int n_in,
                              void* d_out, int out_size, void* d_ws, size_t ws_size,
                              hipStream_t stream) {
  const float* feat   = (const float*)d_in[0];
  const float* fcw    = (const float*)d_in[1];
  const float* emb    = (const float*)d_in[2];
  const int*   e_feat = (const int*)d_in[3];
  const int*   src    = (const int*)d_in[4];
  const int*   dst    = (const int*)d_in[5];

  const int N = in_sizes[0] / IN_F;
  const int E = in_sizes[3];
  const int DB = (out_size - N * OUT_F) / (E * NHEAD);  // drop_blocks (=4)

  float* rst      = (float*)d_out;
  float* attn_out = rst + (size_t)N * OUT_F;

  char* ws = (char*)d_ws;
  float* ft    = (float*)ws;  ws += (size_t)N * OUT_F * sizeof(float);
  int*   cnt   = (int*)ws;    ws += (size_t)N * NETYPE * sizeof(int);
  float* mbuf  = (float*)ws;  ws += (size_t)N * NHEAD * sizeof(float);
  float* invb  = (float*)ws;

  hipMemsetAsync(rst, 0, (size_t)N * OUT_F * sizeof(float), stream);
  hipMemsetAsync(cnt, 0, (size_t)N * NETYPE * sizeof(int), stream);

  const int tiles = N >> 4;                       // 6250 16-row tiles
  srgc_fc_wmma<<<(tiles + 4) / 5, 160, 0, stream>>>(feat, fcw, ft, N);
  srgc_count<<<(E + 255) / 256, 256, 0, stream>>>(e_feat, dst, cnt, E);
  srgc_node_stats<<<(N + 255) / 256, 256, 0, stream>>>(cnt, emb, mbuf, invb, N);
  srgc_edge_attn<<<(E + 255) / 256, 256, 0, stream>>>(e_feat, dst, emb, mbuf,
                                                      invb, attn_out, E, DB);
  srgc_aggregate<<<(E * 32 + 255) / 256, 256, 0, stream>>>(ft, src, dst,
                                                           attn_out, rst, E, DB);
}
